// DECOM_26087631356687
// MI455X (gfx1250) — compile-verified, matrix operand order checked by reading, not run on hardware
//
#include <hip/hip_runtime.h>
#include <hip/hip_bf16.h>

// ---------------------------------------------------------------------------
// Problem constants (fixed by the reference)
// ---------------------------------------------------------------------------
#define BB     8
#define TT     1024
#define DMODEL 1024
#define DFF    4096
#define NH     16
#define DH     64
#define MROWS  (BB * TT)        // 8192
#define NBUCK  33               // 2*CLIP+1

typedef __attribute__((ext_vector_type(16))) _Float16 v16h;
typedef __attribute__((ext_vector_type(8)))  _Float16 v8h;
typedef __attribute__((ext_vector_type(8)))  float    v8f;

union V16 { v16h v; v8h h[2]; };

__device__ __forceinline__ v8f wmma_f16(v16h a, v16h b, v8f c) {
  // D = A(16x32 f16) * B(32x16 f16) + C(16x16 f32)
  return __builtin_amdgcn_wmma_f32_16x16x32_f16(
      /*neg_a=*/false, a, /*neg_b=*/false, b,
      /*c_mod=*/(short)0, c, /*reuse_a=*/false, /*reuse_b=*/false);
}

// --- CDNA5 async global->LDS copy (ASYNCcnt-tracked, no VGPR round trip) ---
__device__ __forceinline__ void async_ld_b128(const void* g, const void* l) {
  // low 32 bits of a generic shared pointer == LDS byte address
  unsigned lds = (unsigned)(unsigned long long)l;
  asm volatile("global_load_async_to_lds_b128 %0, %1, off"
               :: "v"(lds), "v"((unsigned long long)g) : "memory");
}
__device__ __forceinline__ void wait_async0() {
  asm volatile("s_wait_asynccnt 0x0" ::: "memory");
}
__device__ __forceinline__ void wait_ds0() {
  asm volatile("s_wait_dscnt 0x0" ::: "memory");
}

// ---------------------------------------------------------------------------
// f32 -> f16 convert (grid-stride)
// ---------------------------------------------------------------------------
__global__ void cvt_f32_f16_kernel(const float* __restrict__ s,
                                   _Float16* __restrict__ d, int n) {
  for (int i = blockIdx.x * blockDim.x + threadIdx.x; i < n;
       i += gridDim.x * blockDim.x)
    d[i] = (_Float16)s[i];
}

// f32 [K][N] -> f16 transposed [N][K] (weights only; one-shot)
__global__ void cvt_t_f16_kernel(const float* __restrict__ s,
                                 _Float16* __restrict__ d, int K, int N) {
  for (int i = blockIdx.x * blockDim.x + threadIdx.x; i < K * N;
       i += gridDim.x * blockDim.x) {
    const int k = i / N, n = i - k * N;
    d[(size_t)n * K + k] = (_Float16)s[i];
  }
}

// ---------------------------------------------------------------------------
// f16 WMMA GEMM: C[M,N] = A[M,K] * Bt[N,K]^T   (A row-major, B pre-transposed)
// Block tile 128x128, K-step 32, double-buffered LDS, async tile staging.
// 8 waves; each wave computes 32x64 via 2x4 16x16 accumulators (8 WMMA/step).
// ---------------------------------------------------------------------------
#define GSTR 40   // padded LDS stride (halves)

__global__ __launch_bounds__(256) void gemm_f16_kernel(
    const _Float16* __restrict__ A, const _Float16* __restrict__ Bt,
    int M, int N, int K,
    const float* __restrict__ bias, int relu,
    float* __restrict__ outF, _Float16* __restrict__ outH, int splitHeads) {
  __shared__ __align__(16) _Float16 As[2][128 * GSTR];
  __shared__ __align__(16) _Float16 Bs[2][128 * GSTR];

  const int tid  = threadIdx.x;
  const int lane = tid & 31, wid = tid >> 5;
  const int wm   = wid >> 1, wn = wid & 1;          // 4 x 2 wave grid
  const int m0   = blockIdx.y * 128, n0 = blockIdx.x * 128;
  const int mr   = lane & 15, half = lane >> 4, kb = half * 8;

  // each thread stages 2 chunks of A and 2 chunks of B (128 rows x 4 chunks)
  const int r0 = tid >> 2,          ch0 = (tid & 3) * 8;
  const int r1 = (tid + 256) >> 2,  ch1 = (tid & 3) * 8;

  v8f acc[2][4] = {};

  auto issue = [&](int k0, int buf) {
    async_ld_b128(&A [(size_t)(m0 + r0) * K + k0 + ch0], &As[buf][r0 * GSTR + ch0]);
    async_ld_b128(&A [(size_t)(m0 + r1) * K + k0 + ch1], &As[buf][r1 * GSTR + ch1]);
    async_ld_b128(&Bt[(size_t)(n0 + r0) * K + k0 + ch0], &Bs[buf][r0 * GSTR + ch0]);
    async_ld_b128(&Bt[(size_t)(n0 + r1) * K + k0 + ch1], &Bs[buf][r1 * GSTR + ch1]);
  };

  issue(0, 0);
  wait_async0();
  __syncthreads();

  const int nk = K >> 5;
  for (int kt = 0; kt < nk; ++kt) {
    const int cur = kt & 1, nxt = cur ^ 1;
    if (kt + 1 < nk) issue((kt + 1) << 5, nxt);

    V16 bf[4];
#pragma unroll
    for (int ns = 0; ns < 4; ++ns) {
      const int n = wn * 64 + ns * 16 + mr;
      bf[ns].h[0] = *reinterpret_cast<const v8h*>(&Bs[cur][n * GSTR + kb]);
      bf[ns].h[1] = *reinterpret_cast<const v8h*>(&Bs[cur][n * GSTR + kb + 16]);
    }
#pragma unroll
    for (int ms = 0; ms < 2; ++ms) {
      const int m = wm * 32 + ms * 16 + mr;
      V16 af;
      af.h[0] = *reinterpret_cast<const v8h*>(&As[cur][m * GSTR + kb]);
      af.h[1] = *reinterpret_cast<const v8h*>(&As[cur][m * GSTR + kb + 16]);
#pragma unroll
      for (int ns = 0; ns < 4; ++ns)
        acc[ms][ns] = wmma_f16(af.v, bf[ns].v, acc[ms][ns]);
    }

    if (kt + 1 < nk) wait_async0();
    __syncthreads();
  }

#pragma unroll
  for (int ms = 0; ms < 2; ++ms) {
#pragma unroll
    for (int ns = 0; ns < 4; ++ns) {
      const int n = n0 + wn * 64 + ns * 16 + mr;
      const float bv = bias ? bias[n] : 0.0f;
#pragma unroll
      for (int r = 0; r < 8; ++r) {
        const int m = m0 + wm * 32 + ms * 16 + r + half * 8;
        float v = acc[ms][ns][r] + bv;
        if (relu) v = v > 0.0f ? v : 0.0f;
        if (outF) outF[(size_t)m * N + n] = v;
        if (outH) {
          size_t idx;
          if (splitHeads) {
            const int b = m >> 10, t = m & (TT - 1);
            const int hh = n >> 6, d = n & (DH - 1);
            idx = ((size_t)(b * NH + hh) * TT + t) * DH + d;
          } else {
            idx = (size_t)m * N + n;
          }
          outH[idx] = (_Float16)v;
        }
      }
    }
  }
}

// ---------------------------------------------------------------------------
// Flash attention with relative-position key/value bias.
//   Q/K/V: f16 [B*H][T][DH]; ctx: f16 [B][T][DMODEL] (heads merged)
// Block = 256 threads (8 waves); wave handles 16 query rows; 32-key tiles.
// ---------------------------------------------------------------------------
#define KSTR 72
#define VSTR 40

__global__ __launch_bounds__(256) void attn_kernel(
    const _Float16* __restrict__ Qh, const _Float16* __restrict__ Kh,
    const _Float16* __restrict__ Vh,
    const float* __restrict__ posk, const float* __restrict__ posv,
    const int* __restrict__ vlens, _Float16* __restrict__ ctx) {
  __shared__ __align__(16) _Float16 Ks[32 * KSTR];       // [key][d]
  __shared__ __align__(16) _Float16 Vs[64 * VSTR];       // [d][key] (transposed)
  __shared__ __align__(16) _Float16 Pb[8 * 16 * 32];     // per-wave P tiles
  __shared__ float pbuf[8 * 16 * NBUCK];                 // q . pos_k buckets
  __shared__ float bm[8 * 16 * NBUCK];                   // bucket attention mass
  __shared__ float al[8 * 16];                           // per-row alpha bcast
  __shared__ float pvb[NBUCK * DH];                      // pos_v staged

  const int tid = threadIdx.x, lane = tid & 31, wid = tid >> 5;
  const int bh = blockIdx.y, b = bh >> 4, h = bh & 15;
  const int q0 = blockIdx.x * 128 + wid * 16;
  const size_t base = (size_t)bh * TT * DH;
  const int vlen = vlens[b];
  const int kmax = (vlen + 31) & ~31;

  const int mr = lane & 15, half = lane >> 4, kb = half * 8;

  for (int i = tid; i < NBUCK * DH; i += 256) pvb[i] = posv[i];

  // per-wave: pscore[qr][j] = sum_d Q[q0+qr][d] * pos_k[j][d]; zero bucket mass
  for (int idx = lane; idx < 16 * NBUCK; idx += 32) {
    const int qr = idx / NBUCK, j = idx - qr * NBUCK;
    const _Float16* qp = Qh + base + (size_t)(q0 + qr) * DH;
    const float* pk = posk + j * DH;
    float acc = 0.0f;
#pragma unroll 8
    for (int d = 0; d < DH; ++d) acc += (float)qp[d] * pk[d];
    pbuf[wid * (16 * NBUCK) + idx] = acc;
    bm[wid * (16 * NBUCK) + idx] = 0.0f;
  }

  // Q fragments (dh split 0..31 / 32..63)
  V16 qf0, qf1;
  {
    const _Float16* qp = Qh + base + (size_t)(q0 + mr) * DH;
    qf0.h[0] = *reinterpret_cast<const v8h*>(qp + kb);
    qf0.h[1] = *reinterpret_cast<const v8h*>(qp + kb + 16);
    qf1.h[0] = *reinterpret_cast<const v8h*>(qp + 32 + kb);
    qf1.h[1] = *reinterpret_cast<const v8h*>(qp + 32 + kb + 16);
  }

  float m_run[8], l_run[8];
  v8f o0 = {}, o1 = {}, o2 = {}, o3 = {};
#pragma unroll
  for (int r = 0; r < 8; ++r) { m_run[r] = -3.0e38f; l_run[r] = 0.0f; }

  const int ldrow = tid >> 3, ldch = (tid & 7) * 8;

  for (int kt = 0; kt < kmax; kt += 32) {
    __syncthreads();
    // K tile: straight copy -> async DMA to LDS
    async_ld_b128(&Kh[base + (size_t)(kt + ldrow) * DH + ldch],
                  &Ks[ldrow * KSTR + ldch]);
    // V tile: needs transpose -> manual
    v8h vt = *reinterpret_cast<const v8h*>(&Vh[base + (size_t)(kt + ldrow) * DH + ldch]);
#pragma unroll
    for (int c = 0; c < 8; ++c) Vs[(ldch + c) * VSTR + ldrow] = vt[c];
    wait_async0();
    __syncthreads();

    // S = Q * K^T for this 16x32 tile (per wave)
    v8f s0 = {}, s1 = {};
    {
      V16 bk;
      bk.h[0] = *reinterpret_cast<const v8h*>(&Ks[mr * KSTR + kb]);
      bk.h[1] = *reinterpret_cast<const v8h*>(&Ks[mr * KSTR + kb + 16]);
      s0 = wmma_f16(qf0.v, bk.v, s0);
      bk.h[0] = *reinterpret_cast<const v8h*>(&Ks[mr * KSTR + 32 + kb]);
      bk.h[1] = *reinterpret_cast<const v8h*>(&Ks[mr * KSTR + 32 + kb + 16]);
      s0 = wmma_f16(qf1.v, bk.v, s0);
      bk.h[0] = *reinterpret_cast<const v8h*>(&Ks[(16 + mr) * KSTR + kb]);
      bk.h[1] = *reinterpret_cast<const v8h*>(&Ks[(16 + mr) * KSTR + kb + 16]);
      s1 = wmma_f16(qf0.v, bk.v, s1);
      bk.h[0] = *reinterpret_cast<const v8h*>(&Ks[(16 + mr) * KSTR + 32 + kb]);
      bk.h[1] = *reinterpret_cast<const v8h*>(&Ks[(16 + mr) * KSTR + 32 + kb + 16]);
      s1 = wmma_f16(qf1.v, bk.v, s1);
    }

    // online softmax update
    float pe0[8], pe1[8], alpha_r[8];
#pragma unroll
    for (int r = 0; r < 8; ++r) {
      const int qr = half * 8 + r, q = q0 + qr;
      const int k0v = kt + mr, k1v = kt + 16 + mr;
      int d0 = k0v - q; d0 = d0 < -16 ? -16 : (d0 > 16 ? 16 : d0);
      int d1 = k1v - q; d1 = d1 < -16 ? -16 : (d1 > 16 ? 16 : d1);
      const float* pr = &pbuf[wid * (16 * NBUCK) + qr * NBUCK];
      float v0 = (k0v < vlen) ? (s0[r] + pr[d0 + 16]) * 0.125f : -1.0e6f;
      float v1 = (k1v < vlen) ? (s1[r] + pr[d1 + 16]) * 0.125f : -1.0e6f;
      float rm = fmaxf(v0, v1);
      rm = fmaxf(rm, __shfl_xor(rm, 1, 32));
      rm = fmaxf(rm, __shfl_xor(rm, 2, 32));
      rm = fmaxf(rm, __shfl_xor(rm, 4, 32));
      rm = fmaxf(rm, __shfl_xor(rm, 8, 32));
      const float mn = fmaxf(m_run[r], rm);
      const float a = __expf(m_run[r] - mn);
      m_run[r] = mn; alpha_r[r] = a;
      const float p0 = __expf(v0 - mn), p1 = __expf(v1 - mn);
      float rs = p0 + p1;
      rs += __shfl_xor(rs, 1, 32);
      rs += __shfl_xor(rs, 2, 32);
      rs += __shfl_xor(rs, 4, 32);
      rs += __shfl_xor(rs, 8, 32);
      l_run[r] = l_run[r] * a + rs;
      pe0[r] = p0; pe1[r] = p1;
      o0[r] *= a; o1[r] *= a; o2[r] *= a; o3[r] *= a;
    }

    // rescale bucket mass, then add this tile's probabilities
    if (mr == 0) {
#pragma unroll
      for (int r = 0; r < 8; ++r) al[wid * 16 + half * 8 + r] = alpha_r[r];
    }
    wait_ds0();
    if (lane < 16) {
      const float a = al[wid * 16 + lane];
      float* bp = &bm[wid * (16 * NBUCK) + lane * NBUCK];
      for (int j = 0; j < NBUCK; ++j) bp[j] *= a;
    }
    wait_ds0();
#pragma unroll
    for (int r = 0; r < 8; ++r) {
      const int qr = half * 8 + r, q = q0 + qr;
      int d0 = kt + mr - q;      d0 = d0 < -16 ? -16 : (d0 > 16 ? 16 : d0);
      int d1 = kt + 16 + mr - q; d1 = d1 < -16 ? -16 : (d1 > 16 ? 16 : d1);
      atomicAdd(&bm[wid * (16 * NBUCK) + qr * NBUCK + d0 + 16], pe0[r]);
      atomicAdd(&bm[wid * (16 * NBUCK) + qr * NBUCK + d1 + 16], pe1[r]);
    }

    // P (C-layout) -> LDS -> A-fragment, then P * V
#pragma unroll
    for (int r = 0; r < 8; ++r) {
      const int qr = half * 8 + r;
      Pb[(wid * 16 + qr) * 32 + mr]      = (_Float16)pe0[r];
      Pb[(wid * 16 + qr) * 32 + 16 + mr] = (_Float16)pe1[r];
    }
    wait_ds0();
    V16 pf;
    pf.h[0] = *reinterpret_cast<const v8h*>(&Pb[(wid * 16 + mr) * 32 + kb]);
    pf.h[1] = *reinterpret_cast<const v8h*>(&Pb[(wid * 16 + mr) * 32 + kb + 16]);
    V16 vf;
    vf.h[0] = *reinterpret_cast<const v8h*>(&Vs[(0 * 16 + mr) * VSTR + kb]);
    vf.h[1] = *reinterpret_cast<const v8h*>(&Vs[(0 * 16 + mr) * VSTR + kb + 16]);
    o0 = wmma_f16(pf.v, vf.v, o0);
    vf.h[0] = *reinterpret_cast<const v8h*>(&Vs[(1 * 16 + mr) * VSTR + kb]);
    vf.h[1] = *reinterpret_cast<const v8h*>(&Vs[(1 * 16 + mr) * VSTR + kb + 16]);
    o1 = wmma_f16(pf.v, vf.v, o1);
    vf.h[0] = *reinterpret_cast<const v8h*>(&Vs[(2 * 16 + mr) * VSTR + kb]);
    vf.h[1] = *reinterpret_cast<const v8h*>(&Vs[(2 * 16 + mr) * VSTR + kb + 16]);
    o2 = wmma_f16(pf.v, vf.v, o2);
    vf.h[0] = *reinterpret_cast<const v8h*>(&Vs[(3 * 16 + mr) * VSTR + kb]);
    vf.h[1] = *reinterpret_cast<const v8h*>(&Vs[(3 * 16 + mr) * VSTR + kb + 16]);
    o3 = wmma_f16(pf.v, vf.v, o3);
  }

  // epilogue: add bucket-mass * pos_v, normalize, merge heads
  float bmr[NBUCK];
#pragma unroll 1
  for (int r = 0; r < 8; ++r) {
    const int qr = half * 8 + r, q = q0 + qr;
    const float linv = 1.0f / l_run[r];
    for (int j = 0; j < NBUCK; ++j) bmr[j] = bm[wid * (16 * NBUCK) + qr * NBUCK + j];
#pragma unroll
    for (int dk = 0; dk < 4; ++dk) {
      const int d = dk * 16 + mr;
      float extra = 0.0f;
      for (int j = 0; j < NBUCK; ++j) extra += bmr[j] * pvb[j * DH + d];
      const float ov = dk == 0 ? o0[r] : dk == 1 ? o1[r] : dk == 2 ? o2[r] : o3[r];
      const float val = (ov + extra) * linv;
      ctx[((size_t)b * TT + q) * DMODEL + h * DH + d] = (_Float16)val;
    }
  }
}

// ---------------------------------------------------------------------------
// AddNorm: out = LN(X + Dlt) * g + be ; optional f16 copy of out
// ---------------------------------------------------------------------------
__global__ __launch_bounds__(256) void addnorm_kernel(
    const float* __restrict__ X, const float* __restrict__ Dlt,
    const float* __restrict__ g, const float* __restrict__ be,
    float* __restrict__ outF, _Float16* __restrict__ outH) {
  __shared__ float red[2][8];
  const int row = blockIdx.x, tid = threadIdx.x;
  const float* xr = X + (size_t)row * DMODEL;
  const float* dr = Dlt + (size_t)row * DMODEL;
  float v[4], s1 = 0.0f, s2 = 0.0f;
#pragma unroll
  for (int i = 0; i < 4; ++i) {
    const float t = xr[tid + i * 256] + dr[tid + i * 256];
    v[i] = t; s1 += t; s2 += t * t;
  }
  for (int off = 1; off < 32; off <<= 1) {
    s1 += __shfl_xor(s1, off, 32);
    s2 += __shfl_xor(s2, off, 32);
  }
  if ((tid & 31) == 0) { red[0][tid >> 5] = s1; red[1][tid >> 5] = s2; }
  __syncthreads();
  float t1 = 0.0f, t2 = 0.0f;
#pragma unroll
  for (int w = 0; w < 8; ++w) { t1 += red[0][w]; t2 += red[1][w]; }
  const float mu = t1 * (1.0f / DMODEL);
  const float var = t2 * (1.0f / DMODEL) - mu * mu;
  const float rs = rsqrtf(var + 1e-5f);
#pragma unroll
  for (int i = 0; i < 4; ++i) {
    const int c = tid + i * 256;
    const float o = (v[i] - mu) * rs * g[c] + be[c];
    if (outF) outF[(size_t)row * DMODEL + c] = o;
    if (outH) outH[(size_t)row * DMODEL + c] = (_Float16)o;
  }
}

// ---------------------------------------------------------------------------
// Host orchestration
// ---------------------------------------------------------------------------
extern "C" void kernel_launch(void* const* d_in, const int* in_sizes, int n_in,
                              void* d_out, int out_size, void* d_ws, size_t ws_size,
                              hipStream_t stream) {
  (void)in_sizes; (void)n_in; (void)out_size; (void)ws_size;

  const float* x     = (const float*)d_in[0];
  const int*   vlens = (const int*)d_in[1];
  const float* Wq    = (const float*)d_in[2];
  const float* Wk    = (const float*)d_in[3];
  const float* Wv    = (const float*)d_in[4];
  const float* Wo    = (const float*)d_in[5];
  const float* posk  = (const float*)d_in[6];
  const float* posv  = (const float*)d_in[7];
  const float* W1    = (const float*)d_in[8];
  const float* b1    = (const float*)d_in[9];
  const float* W2    = (const float*)d_in[10];
  const float* b2    = (const float*)d_in[11];
  const float* g1    = (const float*)d_in[12];
  const float* be1   = (const float*)d_in[13];
  const float* g2    = (const float*)d_in[14];
  const float* be2   = (const float*)d_in[15];
  float* out = (float*)d_out;

  char* ws = (char*)d_ws;
  size_t off = 0;
  auto alloc = [&](size_t bytes) -> void* {
    void* p = ws + off;
    off += (bytes + 255) & ~(size_t)255;
    return p;
  };

  _Float16* xh   = (_Float16*)alloc((size_t)MROWS * DMODEL * 2);
  _Float16* Wqt  = (_Float16*)alloc((size_t)DMODEL * DMODEL * 2);  // [N][K]
  _Float16* Wkt  = (_Float16*)alloc((size_t)DMODEL * DMODEL * 2);
  _Float16* Wvt  = (_Float16*)alloc((size_t)DMODEL * DMODEL * 2);
  _Float16* Wot  = (_Float16*)alloc((size_t)DMODEL * DMODEL * 2);
  _Float16* W1t  = (_Float16*)alloc((size_t)DMODEL * DFF * 2);     // [DFF][DMODEL]
  _Float16* W2t  = (_Float16*)alloc((size_t)DFF * DMODEL * 2);     // [DMODEL][DFF]
  _Float16* Qh   = (_Float16*)alloc((size_t)MROWS * DMODEL * 2);
  _Float16* Kh   = (_Float16*)alloc((size_t)MROWS * DMODEL * 2);
  _Float16* Vh   = (_Float16*)alloc((size_t)MROWS * DMODEL * 2);
  _Float16* ctxh = (_Float16*)alloc((size_t)MROWS * DMODEL * 2);
  _Float16* yh   = (_Float16*)alloc((size_t)MROWS * DMODEL * 2);
  _Float16* ffnh = (_Float16*)alloc((size_t)MROWS * DFF * 2);
  float*    tmpC = (float*)alloc((size_t)MROWS * DMODEL * 4);
  float*    y    = (float*)alloc((size_t)MROWS * DMODEL * 4);

  // 1) conversions (weights transposed to [N][K] f16)
  cvt_f32_f16_kernel<<<4096, 256, 0, stream>>>(x, xh, MROWS * DMODEL);
  cvt_t_f16_kernel<<<1024, 256, 0, stream>>>(Wq, Wqt, DMODEL, DMODEL);
  cvt_t_f16_kernel<<<1024, 256, 0, stream>>>(Wk, Wkt, DMODEL, DMODEL);
  cvt_t_f16_kernel<<<1024, 256, 0, stream>>>(Wv, Wvt, DMODEL, DMODEL);
  cvt_t_f16_kernel<<<1024, 256, 0, stream>>>(Wo, Wot, DMODEL, DMODEL);
  cvt_t_f16_kernel<<<4096, 256, 0, stream>>>(W1, W1t, DMODEL, DFF);
  cvt_t_f16_kernel<<<4096, 256, 0, stream>>>(W2, W2t, DFF, DMODEL);

  // 2) Q, K, V projections (f16 out, head-split layout)
  dim3 gqkv(DMODEL / 128, MROWS / 128);
  gemm_f16_kernel<<<gqkv, 256, 0, stream>>>(xh, Wqt, MROWS, DMODEL, DMODEL,
                                            nullptr, 0, nullptr, Qh, 1);
  gemm_f16_kernel<<<gqkv, 256, 0, stream>>>(xh, Wkt, MROWS, DMODEL, DMODEL,
                                            nullptr, 0, nullptr, Kh, 1);
  gemm_f16_kernel<<<gqkv, 256, 0, stream>>>(xh, Wvt, MROWS, DMODEL, DMODEL,
                                            nullptr, 0, nullptr, Vh, 1);

  // 3) flash attention with relative-position bias
  attn_kernel<<<dim3(TT / 128, BB * NH), 256, 0, stream>>>(
      Qh, Kh, Vh, posk, posv, vlens, ctxh);

  // 4) output projection -> f32 tmp
  gemm_f16_kernel<<<gqkv, 256, 0, stream>>>(ctxh, Wot, MROWS, DMODEL, DMODEL,
                                            nullptr, 0, tmpC, nullptr, 0);

  // 5) AddNorm 1: y = LN(x + attn_out); also f16 copy
  addnorm_kernel<<<MROWS, 256, 0, stream>>>(x, tmpC, g1, be1, y, yh);

  // 6) FFN layer 1: relu(y @ W1 + b1) -> f16
  gemm_f16_kernel<<<dim3(DFF / 128, MROWS / 128), 256, 0, stream>>>(
      yh, W1t, MROWS, DFF, DMODEL, b1, 1, nullptr, ffnh, 0);

  // 7) FFN layer 2: (ffn @ W2 + b2) -> f32 tmp
  gemm_f16_kernel<<<gqkv, 256, 0, stream>>>(ffnh, W2t, MROWS, DMODEL, DFF,
                                            b2, 0, tmpC, nullptr, 0);

  // 8) AddNorm 2 -> final output (f32)
  addnorm_kernel<<<MROWS, 256, 0, stream>>>(y, tmpC, g2, be2, out, nullptr);
}
